// TGN_46248207843708
// MI455X (gfx1250) — matrix operand order repaired and checked by reading, not compiled, not run on hardware
//
#include <hip/hip_runtime.h>

// ---------------------------------------------------------------------------
// TGN memory-update pipeline for MI455X (gfx1250, wave32, WMMA).
// All GEMMs use v_wmma_f32_16x16x32_bf16 (bf16 in, f32 accumulate).
// d_out doubles as the [N,128] message-sum accumulator before being
// overwritten with the final embeddings (read-before-write per tile).
// ---------------------------------------------------------------------------

typedef __attribute__((ext_vector_type(16))) __bf16 v16bf;
typedef __attribute__((ext_vector_type(8)))  __bf16 v8bf;
typedef __attribute__((ext_vector_type(8)))  float  v8f;

__device__ __forceinline__ __bf16 f2bf(float f) {
  unsigned u = __builtin_bit_cast(unsigned, f);
  u += 0x7FFFu + ((u >> 16) & 1u);          // round-to-nearest-even
  unsigned short h = (unsigned short)(u >> 16);
  return __builtin_bit_cast(__bf16, h);
}

__device__ __forceinline__ v8f wmma_bf16(v16bf a, v16bf b, v8f c) {
  return __builtin_amdgcn_wmma_f32_16x16x32_bf16(
      /*neg_a=*/false, a, /*neg_b=*/false, b,
      /*c_mod=*/(short)0, c, /*reuse_a=*/false, /*reuse_b=*/false);
}

// ------------------------- utility kernels ---------------------------------

__global__ __launch_bounds__(256) void tgn_zero_kernel(float* p, long n) {
  long stride = (long)gridDim.x * 256;
  for (long i = (long)blockIdx.x * 256 + threadIdx.x; i < n; i += stride)
    p[i] = 0.0f;
}

// f32 [rows, scols] -> bf16 [rows, dcols], zero-padding columns >= scols
__global__ __launch_bounds__(256)
void tgn_cvt_kernel(__bf16* dst, const float* src, int rows, int scols, int dcols) {
  long total = (long)rows * dcols;
  long stride = (long)gridDim.x * 256;
  for (long i = (long)blockIdx.x * 256 + threadIdx.x; i < total; i += stride) {
    int r = (int)(i / dcols), c = (int)(i % dcols);
    dst[i] = f2bf(c < scols ? src[(long)r * scols + c] : 0.0f);
  }
}

// ------------------------- message kernel ----------------------------------
// grid = 2 * ceil(B/16); block = 256 (8 wave32).  Tile: 16 edges x 128 msg
// outputs, K = 484 (padded to 512).  dir 0: msg_for_dst (a=src,b=dst, td/
// scatter at dst).  dir 1: msg_for_src (a=dst,b=src, td/scatter at src).

#define XSTR 520   // LDS row stride in bf16 elems (512 + 8 to break conflicts)

__global__ __launch_bounds__(256)
void tgn_msg_kernel(const int* __restrict__ src, const int* __restrict__ dst,
                    const float* __restrict__ ts, const float* __restrict__ efeat,
                    const float* __restrict__ mem, const float* __restrict__ lut,
                    const float* __restrict__ freq, const float* __restrict__ phs,
                    const __bf16* __restrict__ Wb, const float* __restrict__ msgb,
                    float* __restrict__ sums, float* __restrict__ counts, int B) {
  __shared__ __align__(16) __bf16 Xs[16 * XSTR];
  __shared__ int   s_anode[16], s_bnode[16], s_scat[16], s_valid[16];
  __shared__ float s_td[16];

  const int tiles = (B + 15) >> 4;
  const int dir   = (blockIdx.x >= tiles) ? 1 : 0;
  const int tile  = blockIdx.x - dir * tiles;
  const int tid   = threadIdx.x;

  if (tid < 16) {
    int row = tid, edge = tile * 16 + row;
    if (edge < B) {
      int s = src[edge], d = dst[edge];
      int a, b, tnode, scat;
      if (dir == 0) { a = s; b = d; tnode = d; scat = d; }
      else          { a = d; b = s; tnode = s; scat = s; }
      s_anode[row] = a; s_bnode[row] = b; s_scat[row] = scat;
      s_td[row] = ts[edge] - lut[tnode];
      s_valid[row] = 1;
      atomicAdd(&counts[scat], 1.0f);
    } else {
      s_anode[row] = 0; s_bnode[row] = 0; s_scat[row] = 0;
      s_td[row] = 0.0f; s_valid[row] = 0;
    }
  }
  __syncthreads();

  // Stage concatenated input tile [16, 512] as bf16 in LDS.
  for (int e = tid; e < 16 * 512; e += 256) {
    int row = e >> 9, k = e & 511;
    float v = 0.0f;
    if (s_valid[row]) {
      int edge = tile * 16 + row;
      if (k < 128)       v = mem[(long)s_anode[row] * 128 + k];
      else if (k < 256)  v = mem[(long)s_bnode[row] * 128 + (k - 128)];
      else if (k < 356)  { int kk = k - 256; v = cosf(s_td[row] * freq[kk] + phs[kk]); }
      else if (k < 484)  v = efeat[(long)edge * 128 + (k - 356)];
    }
    Xs[row * XSTR + k] = f2bf(v);
  }
  __syncthreads();

  const int wv = tid >> 5, lane = tid & 31;
  const int n = lane & 15, hi = lane >> 4;
  const int jbase = wv * 16;

  v8f acc = {};
#pragma unroll 4
  for (int kt = 0; kt < 16; ++kt) {
    // A frag (16x32 bf16): lane row = n; K = kt*32 + 16*hi .. +15 contiguous.
    const __bf16* ap = &Xs[n * XSTR + kt * 32 + hi * 16];
    v8bf lo = *(const v8bf*)ap;
    v8bf h8 = *(const v8bf*)(ap + 8);
    v16bf a = __builtin_shufflevector(lo, h8, 0,1,2,3,4,5,6,7,8,9,10,11,12,13,14,15);
    // B frag (32x16): lane col = n; K = kt*32 + 16*hi .. +15 contiguous.
    v16bf b = *(const v16bf*)(Wb + (long)(jbase + n) * 512 + kt * 32 + hi * 16);
    acc = wmma_bf16(a, b, acc);
  }

  // Epilogue: ReLU(bias+acc), scatter-add into sums.
  float bj = msgb[jbase + n];
#pragma unroll
  for (int i = 0; i < 8; ++i) {
    int m = hi * 8 + i;                      // C/D layout: VGPR i -> M = i + 8*hi
    if (s_valid[m]) {
      float v = acc[i] + bj;
      v = v > 0.0f ? v : 0.0f;
      atomicAdd(&sums[(long)s_scat[m] * 128 + jbase + n], v);
    }
  }
}

// ------------------------- node update kernel ------------------------------
// grid = ceil(N/16); block = 256.  Per tile of 16 nodes:
//   agg = sums/max(cnt,1) ; gi = agg@Wih^T ; gh = mem@Whh^T ; GRU gates ;
//   new_mem ; emb = new_mem@Wemb^T + b -> overwrites sums rows in d_out.

__global__ __launch_bounds__(256)
void tgn_upd_kernel(const float* __restrict__ mem, const float* __restrict__ counts,
                    const float* __restrict__ bih, const float* __restrict__ bhh,
                    const __bf16* __restrict__ Wih, const __bf16* __restrict__ Whh,
                    const __bf16* __restrict__ Wemb, const float* __restrict__ embb,
                    float* __restrict__ out, int N) {
  __shared__ __align__(16) float agg_f[16 * 132];   // later reused as new_mem
  __shared__ __align__(16) float mem_f[16 * 132];
  __shared__ float gi_s[16 * 388];
  __shared__ float gh_s[16 * 388];
  __shared__ float s_cnt[16];

  const int n0 = blockIdx.x * 16;
  const int tid = threadIdx.x;

  if (tid < 16) {
    int node = n0 + tid;
    s_cnt[tid] = (node < N) ? counts[node] : 0.0f;
  }
  __syncthreads();

  for (int e = tid; e < 2048; e += 256) {
    int r = e >> 7, c = e & 127, node = n0 + r;
    float sv = 0.0f, mv = 0.0f;
    if (node < N) { sv = out[(long)node * 128 + c]; mv = mem[(long)node * 128 + c]; }
    agg_f[r * 132 + c] = sv / fmaxf(s_cnt[r], 1.0f);
    mem_f[r * 132 + c] = mv;
  }
  __syncthreads();

  const int wv = tid >> 5, lane = tid & 31;
  const int n = lane & 15, hi = lane >> 4;

  // 48 tile-jobs: 24 gi N-tiles (384 cols) + 24 gh N-tiles; 6 per wave.
  for (int q = 0; q < 6; ++q) {
    int job = wv + 8 * q;
    bool is_gh = job >= 24;
    int ntile = is_gh ? job - 24 : job;
    const float*  A  = is_gh ? mem_f : agg_f;
    const __bf16* Bw = is_gh ? Whh : Wih;
    v8f acc = {};
#pragma unroll
    for (int kt = 0; kt < 4; ++kt) {
      const float* ap = &A[n * 132 + kt * 32 + hi * 16];
      v16bf a;
#pragma unroll
      for (int i = 0; i < 8; ++i) { a[i] = f2bf(ap[i]); a[i + 8] = f2bf(ap[i + 8]); }
      v16bf b = *(const v16bf*)(Bw + (long)(ntile * 16 + n) * 128 + kt * 32 + hi * 16);
      acc = wmma_bf16(a, b, acc);
    }
    float* G = is_gh ? gh_s : gi_s;
#pragma unroll
    for (int i = 0; i < 8; ++i)
      G[(hi * 8 + i) * 388 + ntile * 16 + n] = acc[i];
  }
  __syncthreads();

  // GRU gate math; new_mem written into agg_f (safe after barrier).
  for (int e = tid; e < 2048; e += 256) {
    int r = e >> 7, h = e & 127;
    float gir = gi_s[r * 388 + h]       + bih[h];
    float giz = gi_s[r * 388 + 128 + h] + bih[128 + h];
    float gin = gi_s[r * 388 + 256 + h] + bih[256 + h];
    float ghr = gh_s[r * 388 + h]       + bhh[h];
    float ghz = gh_s[r * 388 + 128 + h] + bhh[128 + h];
    float ghn = gh_s[r * 388 + 256 + h] + bhh[256 + h];
    float rr = 1.0f / (1.0f + expf(-(gir + ghr)));
    float zz = 1.0f / (1.0f + expf(-(giz + ghz)));
    float nn = tanhf(gin + rr * ghn);
    float hv = mem_f[r * 132 + h];
    float up = (1.0f - zz) * nn + zz * hv;
    agg_f[r * 132 + h] = (s_cnt[r] > 0.0f) ? up : hv;
  }
  __syncthreads();

  // Embedding GEMM: wave wv owns output N-tile wv (128 cols total).
  v8f acc = {};
#pragma unroll
  for (int kt = 0; kt < 4; ++kt) {
    const float* ap = &agg_f[n * 132 + kt * 32 + hi * 16];
    v16bf a;
#pragma unroll
    for (int i = 0; i < 8; ++i) { a[i] = f2bf(ap[i]); a[i + 8] = f2bf(ap[i + 8]); }
    v16bf b = *(const v16bf*)(Wemb + (long)(wv * 16 + n) * 128 + kt * 32 + hi * 16);
    acc = wmma_bf16(a, b, acc);
  }
  float eb = embb[wv * 16 + n];
#pragma unroll
  for (int i = 0; i < 8; ++i) {
    int node = n0 + hi * 8 + i;
    if (node < N) out[(long)node * 128 + wv * 16 + n] = acc[i] + eb;
  }
}

// ------------------------- host launch -------------------------------------

extern "C" void kernel_launch(void* const* d_in, const int* in_sizes, int n_in,
                              void* d_out, int out_size, void* d_ws, size_t ws_size,
                              hipStream_t stream) {
  const int*   srcn  = (const int*)d_in[0];
  const int*   dstn  = (const int*)d_in[1];
  const float* ts    = (const float*)d_in[2];
  const float* efeat = (const float*)d_in[3];
  const float* mem   = (const float*)d_in[4];
  const float* lut   = (const float*)d_in[5];
  const float* freq  = (const float*)d_in[6];
  const float* phs   = (const float*)d_in[7];
  const float* msgW  = (const float*)d_in[8];
  const float* msgb  = (const float*)d_in[9];
  const float* wih   = (const float*)d_in[10];
  const float* whh   = (const float*)d_in[11];
  const float* bih   = (const float*)d_in[12];
  const float* bhh   = (const float*)d_in[13];
  const float* embW  = (const float*)d_in[14];
  const float* embb  = (const float*)d_in[15];

  const int B = in_sizes[0];      // 100000
  const int N = in_sizes[5];      // 200000
  float* out = (float*)d_out;     // sums accumulator, then embeddings

  // workspace layout (≈1.2 MB)
  char* ws = (char*)d_ws;
  __bf16* Wmsg_bf = (__bf16*)(ws + 0);        // 128 x 512
  __bf16* Wih_bf  = (__bf16*)(ws + 131072);   // 384 x 128
  __bf16* Whh_bf  = (__bf16*)(ws + 229376);   // 384 x 128
  __bf16* Wemb_bf = (__bf16*)(ws + 327680);   // 128 x 128
  float*  counts  = (float*)(ws + 360448);    // N

  tgn_zero_kernel<<<2048, 256, 0, stream>>>(out, (long)N * 128);
  tgn_zero_kernel<<<256, 256, 0, stream>>>(counts, (long)N);
  tgn_cvt_kernel<<<256, 256, 0, stream>>>(Wmsg_bf, msgW, 128, 484, 512);
  tgn_cvt_kernel<<<192, 256, 0, stream>>>(Wih_bf,  wih,  384, 128, 128);
  tgn_cvt_kernel<<<192, 256, 0, stream>>>(Whh_bf,  whh,  384, 128, 128);
  tgn_cvt_kernel<<<64, 256, 0, stream>>>(Wemb_bf, embW, 128, 128, 128);

  int tiles = (B + 15) / 16;
  tgn_msg_kernel<<<2 * tiles, 256, 0, stream>>>(
      srcn, dstn, ts, efeat, mem, lut, freq, phs, Wmsg_bf, msgb, out, counts, B);
  tgn_upd_kernel<<<(N + 15) / 16, 256, 0, stream>>>(
      mem, counts, bih, bhh, Wih_bf, Whh_bf, Wemb_bf, embb, out, N);
}
